// MultiheadAtt_58557584114084
// MI455X (gfx1250) — matrix operand order repaired and checked by reading, not compiled, not run on hardware
//
#include <hip/hip_runtime.h>
#include <hip/hip_bf16.h>

// ---------------------------------------------------------------------------
// CDNA5 (gfx1250) multi-head attention forward.
// bf16 WMMA (v_wmma_f32_16x16x32_bf16), f32 accumulate, double-buffered LDS.
// bf16 operand staging uses GLOBAL_LOAD_ASYNC_TO_LDS_B128 (ASYNCcnt) when the
// toolchain exposes it; f32 operands use register-staged convert copies.
// ---------------------------------------------------------------------------

typedef __attribute__((ext_vector_type(16))) __bf16 v16bf;
typedef __attribute__((ext_vector_type(8)))  float  v8f;
typedef __attribute__((ext_vector_type(4)))  unsigned int u32x4;
typedef __attribute__((ext_vector_type(4)))  int v4i;

#if defined(__has_builtin)
#  if __has_builtin(__builtin_amdgcn_global_load_async_to_lds_b128)
#    define HAVE_ASYNC_LDS 1
#  endif
#endif
#ifndef HAVE_ASYNC_LDS
#  define HAVE_ASYNC_LDS 0
#endif

#if HAVE_ASYNC_LDS
typedef __attribute__((address_space(1))) v4i* gv4i_ptr;
typedef __attribute__((address_space(3))) v4i* lv4i_ptr;
__device__ __forceinline__ void async_copy_b128(const __bf16* g, __bf16* l) {
  __builtin_amdgcn_global_load_async_to_lds_b128(
      (gv4i_ptr)g, (lv4i_ptr)l, /*offset=*/0, /*cpol=*/0);
}
__device__ __forceinline__ void wait_async0() {
#  if __has_builtin(__builtin_amdgcn_s_wait_asynccnt)
  __builtin_amdgcn_s_wait_asynccnt(0);
#  else
  asm volatile("s_wait_asynccnt 0x0" ::: "memory");
#  endif
}
#else
__device__ __forceinline__ void wait_async0() {}
#endif

union FragU { v16bf v; u32x4 q[2]; };

__device__ __forceinline__ void store_elem(float* p, float v)  { *p = v; }
__device__ __forceinline__ void store_elem(__bf16* p, float v) { *p = (__bf16)v; }

// Raw 8-element chunk staged through registers; conversion deferred to the
// LDS-store phase so global loads can stay in flight behind the WMMAs.
template <typename T> struct RawChunk;
template <> struct RawChunk<float> {
  float4 a, b;
  __device__ __forceinline__ void load(const float* p) {
    a = *(const float4*)p; b = *(const float4*)(p + 4);
  }
  __device__ __forceinline__ u32x4 to_bf16x8() const {
    union { __bf16 h[8]; u32x4 r; } u;
    u.h[0] = (__bf16)a.x; u.h[1] = (__bf16)a.y; u.h[2] = (__bf16)a.z; u.h[3] = (__bf16)a.w;
    u.h[4] = (__bf16)b.x; u.h[5] = (__bf16)b.y; u.h[6] = (__bf16)b.z; u.h[7] = (__bf16)b.w;
    return u.r;
  }
};
template <> struct RawChunk<__bf16> {
  u32x4 a;
  __device__ __forceinline__ void load(const __bf16* p) { a = *(const u32x4*)p; }
  __device__ __forceinline__ u32x4 to_bf16x8() const { return a; }
};

// A fragment (16x32 bf16): lanes 0-15 hold row M=lane, K {0..7,16..23};
// lanes 16-31 hold row M=lane-16, K {8..15,24..31}. LDS tile is row-major.
__device__ __forceinline__ v16bf load_frag_a(const __bf16* s, int lda, int lane) {
  const int half = lane >> 4, r = lane & 15;
  const __bf16* p = s + r * lda + half * 8;
  FragU u;
  u.q[0] = *(const u32x4*)(p);
  u.q[1] = *(const u32x4*)(p + 16);
  return u.v;
}
// B fragment (32x16 bf16): lane = column N; lanes 0-15 hold K {0..7,16..23},
// lanes 16-31 hold K {8..15,24..31}. LDS tile stored as [N][K] row-major.
__device__ __forceinline__ v16bf load_frag_b(const __bf16* s, int ldb, int lane) {
  const int half = lane >> 4, c = lane & 15;
  const __bf16* p = s + c * ldb + half * 8;
  FragU u;
  u.q[0] = *(const u32x4*)(p);
  u.q[1] = *(const u32x4*)(p + 16);
  return u.v;
}

__device__ __forceinline__ v8f wmma_bf16(v16bf a, v16bf b, v8f c) {
  return __builtin_amdgcn_wmma_f32_16x16x32_bf16(
      false, a, false, b, (short)0, c, false, false);
}

// ---------------------------------------------------------------------------
// Batched transpose+convert:  Y[z][c][r] = (bf16) X[z][r][c]
// 64x64 tiles, 256 threads; vector global I/O, padded LDS tile.
// Requires R%64==0, C%64==0.
// ---------------------------------------------------------------------------
template <typename TIN>
__global__ void __launch_bounds__(256)
transpose_bf16_kernel(const TIN* __restrict__ X, long long strideX,
                      __bf16* __restrict__ Y, long long strideY,
                      int R, int C) {
  __shared__ __align__(16) __bf16 t[64][72];
  X += (long long)blockIdx.z * strideX;
  Y += (long long)blockIdx.z * strideY;
  const int tid = threadIdx.x;
  const int r0 = blockIdx.y * 64, c0 = blockIdx.x * 64;

  RawChunk<TIN> rc[2];
  #pragma unroll
  for (int s = 0; s < 2; ++s) {
    const int i = tid + s * 256, r = i >> 3, c8 = (i & 7) << 3;
    rc[s].load(&X[(long long)(r0 + r) * C + c0 + c8]);
  }
  #pragma unroll
  for (int s = 0; s < 2; ++s) {
    const int i = tid + s * 256, r = i >> 3, c8 = (i & 7) << 3;
    *(u32x4*)&t[r][c8] = rc[s].to_bf16x8();
  }
  __syncthreads();
  #pragma unroll
  for (int s = 0; s < 2; ++s) {
    const int i = tid + s * 256, c = i >> 3, r8 = (i & 7) << 3;
    union { __bf16 h[8]; u32x4 r; } u;
    #pragma unroll
    for (int j = 0; j < 8; ++j) u.h[j] = t[r8 + j][c];
    *(u32x4*)&Y[(long long)(c0 + c) * R + r0 + r8] = u.r;
  }
}

// ---------------------------------------------------------------------------
// Batched GEMM:  Y[z] = alpha * X[z] @ Bt[z]^T + bias,  Bt stored bf16 [N][K]
// 256 threads = 8 wave32; wave grid (BM/WM)x(BN/WN) must equal 8.
// Double-buffered LDS, one barrier per 32-deep K-step.
// Requires M%BM==0, N%BN==0, K%32==0.
// ---------------------------------------------------------------------------
template <typename TIN, typename TOUT, int BM, int BN, int WM, int WN>
__global__ void __launch_bounds__(256)
gemm_tile_kernel(const TIN* __restrict__ X, long long strideX,
                 const __bf16* __restrict__ Bt, long long strideB,
                 const float* __restrict__ bias,
                 TOUT* __restrict__ Y, long long strideY,
                 int M, int N, int K, float alpha) {
  constexpr int BK = 32, LDT = BK + 8;             // 80 B rows: keeps 16B align
  constexpr int MI = WM / 16, NI = WN / 16;
  constexpr int WAVES_N = BN / WN;
  constexpr int PA = (BM * 4) / 256;               // 8-elem A chunks / thread
  constexpr int PB = (BN * 4) / 256;               // 8-elem B chunks / thread
  constexpr bool ASYNC_A = HAVE_ASYNC_LDS && __is_same(TIN, __bf16);
  constexpr bool ASYNC_B = HAVE_ASYNC_LDS;         // Bt is always bf16
  static_assert((BM / WM) * (BN / WN) == 8, "8 waves");

  __shared__ __align__(16) __bf16 As[2][BM * LDT];
  __shared__ __align__(16) __bf16 Bs[2][BN * LDT];

  X  += (long long)blockIdx.z * strideX;
  Bt += (long long)blockIdx.z * strideB;
  Y  += (long long)blockIdx.z * strideY;

  const int tid = threadIdx.x, lane = tid & 31, wave = tid >> 5;
  const int wm = wave / WAVES_N, wn = wave % WAVES_N;
  const int m0 = blockIdx.y * BM, n0 = blockIdx.x * BN;

  RawChunk<TIN>    ra[PA];
  RawChunk<__bf16> rb[PB];

  auto loadA = [&](int k0) {
    #pragma unroll
    for (int t = 0; t < PA; ++t) {
      const int i = tid + t * 256, r = i >> 2, c8 = (i & 3) << 3;
      ra[t].load(&X[(long long)(m0 + r) * K + k0 + c8]);
    }
  };
  auto storeA = [&](__bf16* buf) {
    #pragma unroll
    for (int t = 0; t < PA; ++t) {
      const int i = tid + t * 256, r = i >> 2, c8 = (i & 3) << 3;
      *(u32x4*)&buf[r * LDT + c8] = ra[t].to_bf16x8();
    }
  };
  auto loadB = [&](int k0) {
    #pragma unroll
    for (int t = 0; t < PB; ++t) {
      const int i = tid + t * 256, n = i >> 2, c8 = (i & 3) << 3;
      rb[t].load(&Bt[(long long)(n0 + n) * K + k0 + c8]);
    }
  };
  auto storeB = [&](__bf16* buf) {
    #pragma unroll
    for (int t = 0; t < PB; ++t) {
      const int i = tid + t * 256, n = i >> 2, c8 = (i & 3) << 3;
      *(u32x4*)&buf[n * LDT + c8] = rb[t].to_bf16x8();
    }
  };
#if HAVE_ASYNC_LDS
  auto stageA_async = [&](int k0, __bf16* buf) {
    #pragma unroll
    for (int t = 0; t < PA; ++t) {
      const int i = tid + t * 256, r = i >> 2, c8 = (i & 3) << 3;
      async_copy_b128((const __bf16*)&X[(long long)(m0 + r) * K + k0 + c8],
                      &buf[r * LDT + c8]);
    }
  };
  auto stageB_async = [&](int k0, __bf16* buf) {
    #pragma unroll
    for (int t = 0; t < PB; ++t) {
      const int i = tid + t * 256, n = i >> 2, c8 = (i & 3) << 3;
      async_copy_b128(&Bt[(long long)(n0 + n) * K + k0 + c8],
                      &buf[n * LDT + c8]);
    }
  };
#endif

  v8f acc[MI][NI];
  #pragma unroll
  for (int mi = 0; mi < MI; ++mi)
    #pragma unroll
    for (int ni = 0; ni < NI; ++ni) acc[mi][ni] = {};

  const int nsteps = K / BK;

  // prologue: stage tile 0
#if HAVE_ASYNC_LDS
  if constexpr (ASYNC_A) stageA_async(0, As[0]); else { loadA(0); storeA(As[0]); }
  if constexpr (ASYNC_B) stageB_async(0, Bs[0]); else { loadB(0); storeB(Bs[0]); }
  wait_async0();
#else
  loadA(0); loadB(0);
  storeA(As[0]); storeB(Bs[0]);
#endif
  __syncthreads();

  for (int s = 0; s < nsteps; ++s) {
    const int cur = s & 1;
    const bool more = (s + 1 < nsteps);
    // issue next tile's loads before compute (kept in flight behind WMMAs)
    if (more) {
#if HAVE_ASYNC_LDS
      if constexpr (ASYNC_A) stageA_async((s + 1) * BK, As[cur ^ 1]); else loadA((s + 1) * BK);
      if constexpr (ASYNC_B) stageB_async((s + 1) * BK, Bs[cur ^ 1]); else loadB((s + 1) * BK);
#else
      loadA((s + 1) * BK); loadB((s + 1) * BK);
#endif
    }

    v16bf a[MI], b[NI];
    #pragma unroll
    for (int mi = 0; mi < MI; ++mi)
      a[mi] = load_frag_a(&As[cur][(wm * WM + mi * 16) * LDT], LDT, lane);
    #pragma unroll
    for (int ni = 0; ni < NI; ++ni)
      b[ni] = load_frag_b(&Bs[cur][(wn * WN + ni * 16) * LDT], LDT, lane);
    #pragma unroll
    for (int mi = 0; mi < MI; ++mi)
      #pragma unroll
      for (int ni = 0; ni < NI; ++ni)
        acc[mi][ni] = wmma_bf16(a[mi], b[ni], acc[mi][ni]);

    // drain register-staged copies, then synchronize the tile swap
    if (more) {
      if constexpr (!ASYNC_A) storeA(As[cur ^ 1]);
      if constexpr (!ASYNC_B) storeB(Bs[cur ^ 1]);
      if constexpr (ASYNC_A || ASYNC_B) wait_async0();
    }
    __syncthreads();
  }

  // epilogue: C/D layout VGPR j -> row (lane>=16?8:0)+j, col lane&15
  const int rl = (lane >> 4) * 8, cl = lane & 15;
  #pragma unroll
  for (int mi = 0; mi < MI; ++mi) {
    #pragma unroll
    for (int ni = 0; ni < NI; ++ni) {
      const long long row0 = m0 + wm * WM + mi * 16 + rl;
      const int col = n0 + wn * WN + ni * 16 + cl;
      const float bv = bias ? bias[col] : 0.0f;
      #pragma unroll
      for (int j = 0; j < 8; ++j)
        store_elem(&Y[(row0 + j) * (long long)N + col], acc[mi][ni][j] * alpha + bv);
    }
  }
}

// ---------------------------------------------------------------------------
// Row softmax, in place. One 256-thread block (8 wave32) per row of L=2048.
// ---------------------------------------------------------------------------
__global__ void __launch_bounds__(256)
softmax_rows_kernel(float* __restrict__ probs, int L) {
  float* p = probs + (long long)blockIdx.x * L;
  const int tid = threadIdx.x, lane = tid & 31, wave = tid >> 5;
  __shared__ float sm[8], ss[8];

  float4* pv = (float4*)(p + tid * 8);
  float4 x0 = pv[0], x1 = pv[1];
  float v[8] = {x0.x, x0.y, x0.z, x0.w, x1.x, x1.y, x1.z, x1.w};

  float m = v[0];
  #pragma unroll
  for (int j = 1; j < 8; ++j) m = fmaxf(m, v[j]);
  #pragma unroll
  for (int o = 16; o > 0; o >>= 1) m = fmaxf(m, __shfl_xor(m, o, 32));
  if (lane == 0) sm[wave] = m;
  __syncthreads();
  float rmax = sm[0];
  #pragma unroll
  for (int i = 1; i < 8; ++i) rmax = fmaxf(rmax, sm[i]);

  float s = 0.0f;
  #pragma unroll
  for (int j = 0; j < 8; ++j) { v[j] = __expf(v[j] - rmax); s += v[j]; }
  #pragma unroll
  for (int o = 16; o > 0; o >>= 1) s += __shfl_xor(s, o, 32);
  if (lane == 0) ss[wave] = s;
  __syncthreads();
  float tot = 0.0f;
  #pragma unroll
  for (int i = 0; i < 8; ++i) tot += ss[i];
  const float inv = 1.0f / tot;

  x0.x = v[0] * inv; x0.y = v[1] * inv; x0.z = v[2] * inv; x0.w = v[3] * inv;
  x1.x = v[4] * inv; x1.y = v[5] * inv; x1.z = v[6] * inv; x1.w = v[7] * inv;
  pv[0] = x0; pv[1] = x1;
}

// ---------------------------------------------------------------------------
extern "C" void kernel_launch(void* const* d_in, const int* in_sizes, int n_in,
                              void* d_out, int out_size, void* d_ws, size_t ws_size,
                              hipStream_t stream) {
  (void)in_sizes; (void)n_in; (void)out_size; (void)ws_size;
  constexpr int B = 4, S = 2048, D = 512, H = 8, HD = 64;
  constexpr int M = B * S;                           // 8192
  constexpr long long QKV = (long long)B * S * D;    // 4,194,304 elems
  constexpr int G = B * H;                           // 32 slabs of [2048][64]
  constexpr long long DD = (long long)D * D;         // 262,144 elems

  const float* query = (const float*)d_in[0];
  const float* key_  = (const float*)d_in[1];
  const float* value = (const float*)d_in[2];
  const float* Wq = (const float*)d_in[3];  const float* bq = (const float*)d_in[4];
  const float* Wk = (const float*)d_in[5];  const float* bk = (const float*)d_in[6];
  const float* Wv = (const float*)d_in[7];  const float* bv = (const float*)d_in[8];
  const float* Wo = (const float*)d_in[9];  const float* bo = (const float*)d_in[10];

  // workspace (all bf16): activations + transposed operands
  __bf16* q_ws    = (__bf16*)d_ws;
  __bf16* k_ws    = q_ws + QKV;
  __bf16* v_ws    = k_ws + QKV;
  __bf16* attn_ws = v_ws + QKV;
  __bf16* v_t     = attn_ws + QKV;     // [G][64][2048]
  __bf16* Wq_t    = v_t + QKV;         // [512][512] (N-major)
  __bf16* Wk_t    = Wq_t + DD;
  __bf16* Wv_t    = Wk_t + DD;
  __bf16* Wo_t    = Wv_t + DD;

  float* out_  = (float*)d_out;          // [B][S][D]
  float* probs = out_ + QKV;             // [G][S][S]

  const dim3 blk(256);

  // 0) one-time: weights f32 [K][N] -> bf16 [N][K]
  const dim3 gtw(D / 64, D / 64, 1);
  transpose_bf16_kernel<float><<<gtw, blk, 0, stream>>>(Wq, 0, Wq_t, 0, D, D);
  transpose_bf16_kernel<float><<<gtw, blk, 0, stream>>>(Wk, 0, Wk_t, 0, D, D);
  transpose_bf16_kernel<float><<<gtw, blk, 0, stream>>>(Wv, 0, Wv_t, 0, D, D);
  transpose_bf16_kernel<float><<<gtw, blk, 0, stream>>>(Wo, 0, Wo_t, 0, D, D);

  // 1-3) projections: f32 @ bf16^T + bias -> bf16  (M=8192, N=512, K=512)
  const dim3 gproj(D / 128, M / 128, 1);
  gemm_tile_kernel<float, __bf16, 128, 128, 32, 64>
      <<<gproj, blk, 0, stream>>>(query, 0, Wq_t, 0, bq, q_ws, 0, M, D, D, 1.0f);
  gemm_tile_kernel<float, __bf16, 128, 128, 32, 64>
      <<<gproj, blk, 0, stream>>>(key_, 0, Wk_t, 0, bk, k_ws, 0, M, D, D, 1.0f);
  gemm_tile_kernel<float, __bf16, 128, 128, 32, 64>
      <<<gproj, blk, 0, stream>>>(value, 0, Wv_t, 0, bv, v_ws, 0, M, D, D, 1.0f);

  // 3b) one-time: v slabs [2048][64] -> v_t slabs [64][2048]
  const dim3 gtv(1, S / 64, G);
  transpose_bf16_kernel<__bf16><<<gtv, blk, 0, stream>>>(
      v_ws, (long long)S * HD, v_t, (long long)S * HD, S, HD);

  // 4) scores = (q @ k^T) / 8  per slab  (M=N=2048, K=64) -> probs f32
  const dim3 gsc(S / 128, S / 128, G);
  gemm_tile_kernel<__bf16, float, 128, 128, 32, 64>
      <<<gsc, blk, 0, stream>>>(q_ws, (long long)S * HD, k_ws, (long long)S * HD,
                                nullptr, probs, (long long)S * S, S, S, HD, 0.125f);

  // 5) softmax over last dim, in place in probs
  softmax_rows_kernel<<<dim3(G * S), blk, 0, stream>>>(probs, S);

  // 6) attn = probs @ v  per slab  (M=2048, N=64, K=2048) -> bf16
  const dim3 gpv(HD / 64, S / 128, G);
  gemm_tile_kernel<float, __bf16, 128, 64, 32, 32>
      <<<gpv, blk, 0, stream>>>(probs, (long long)S * S, v_t, (long long)S * HD,
                                nullptr, attn_ws, (long long)S * HD, S, HD, S, 1.0f);

  // 7) out = attn @ Wo + bo   (flat reinterpret [8192][512]) -> f32 d_out
  gemm_tile_kernel<__bf16, float, 128, 128, 32, 64>
      <<<gproj, blk, 0, stream>>>(attn_ws, 0, Wo_t, 0, bo, out_, 0, M, D, D, 1.0f);
}
